// FlashAttention_53094385713646
// MI455X (gfx1250) — compile-verified, matrix operand order checked by reading, not run on hardware
//
#include <hip/hip_runtime.h>
#include <hip/hip_bf16.h>
#include <math.h>

// ---------------------------------------------------------------------------
// Multi-head attention (B=2, S=2048, D=1024, H=16, DK=64) for gfx1250.
// f32->f16 convert -> WMMA QKV projections -> wave32 flash attention computed
// TRANSPOSED (S^T = K Q^T, O^T = V^T P^T) so softmax is per-lane and the
// probabilities are already in WMMA B-operand layout (no LDS shuffle, no
// cross-lane butterflies) -> WMMA output projection (f32 out).
// ---------------------------------------------------------------------------

typedef __attribute__((ext_vector_type(16))) _Float16 v16h;
typedef __attribute__((ext_vector_type(8)))  _Float16 v8h;
typedef __attribute__((ext_vector_type(4)))  _Float16 v4h;
typedef __attribute__((ext_vector_type(8)))  float    v8f;

#define DDIM 1024
#define SEQ  2048
#define NB   2
#define NH   16
#define DKD  64
#define MTOT (NB * SEQ)   // 4096 tokens

__device__ __forceinline__ v8f wmma_f32_f16(v16h a, v16h b, v8f c) {
  // (neg_a, A, neg_b, B, c_mod, C, reuse_a, reuse_b)
  return __builtin_amdgcn_wmma_f32_16x16x32_f16(false, a, false, b,
                                                (short)0, c, false, false);
}

// A/B fragment: lanes 0-15 hold row (M) / col (N) = lane&15 with K elems
// {0..7,16..23}; lanes 16-31 hold the same row/col with K elems {8..15,24..31}.
// Element e <-> K = (e>=8?16:0) + 8*(lane>=16) + (e&7).
// Two contiguous 16-byte (8-half) loads per lane.
__device__ __forceinline__ v16h load_frag(const _Float16* p0, const _Float16* p1) {
  v8h lo = *(const v8h*)p0;
  v8h hi = *(const v8h*)p1;
  v16h r;
#pragma unroll
  for (int i = 0; i < 8; ++i) { r[i] = lo[i]; r[8 + i] = hi[i]; }
  return r;
}

// ---------------------------------------------------------------------------
// fp32 -> f16 conversion (vectorized, grid-stride). n4 = n/4.
// ---------------------------------------------------------------------------
__global__ void cvt_f32_f16(const float* __restrict__ src,
                            _Float16* __restrict__ dst, int n4) {
  int i = blockIdx.x * blockDim.x + threadIdx.x;
  int stride = gridDim.x * blockDim.x;
  for (; i < n4; i += stride) {
    float4 v = ((const float4*)src)[i];
    v4h h;
    h[0] = (_Float16)v.x; h[1] = (_Float16)v.y;
    h[2] = (_Float16)v.z; h[3] = (_Float16)v.w;
    ((v4h*)dst)[i] = h;
  }
}

// ---------------------------------------------------------------------------
// WMMA GEMM: out[t][e] = sum_k A[t][k] * W[e][k] + bias[e]   (torch x @ W^T + b)
// A: MTOT x 1024 f16 row-major. W: 1024 x 1024 f16 row-major (rows = out feat).
// Block = 128 threads (4 waves), each wave owns a 16x64 output tile.
// MODE 0: Q  -> f16 [B,H,S,DK], scaled by 1/sqrt(DK)
// MODE 1: K  -> f16 [B,H,S,DK]
// MODE 2: V  -> f16 [B,H,DK,S]  (transposed: A-operand of O^T = V^T P^T)
// MODE 3: final output -> f32 [t][e]
// ---------------------------------------------------------------------------
template <int MODE>
__global__ __launch_bounds__(128) void gemm_wmma(const _Float16* __restrict__ A,
                                                 const _Float16* __restrict__ W,
                                                 const float* __restrict__ bias,
                                                 void* __restrict__ out) {
  const int lane = threadIdx.x & 31;
  const int wave = threadIdx.x >> 5;
  const int halfw = lane >> 4;
  const int ln   = lane & 15;
  const int koff = halfw * 8;
  const int r0 = blockIdx.x * 64 + wave * 16;   // token tile
  const int c0 = blockIdx.y * 64;               // feature tile

  const _Float16* arow = A + (size_t)(r0 + ln) * DDIM;
  v8f acc[4] = {};

  for (int k0 = 0; k0 < DDIM; k0 += 32) {
    v16h a = load_frag(arow + k0 + koff, arow + k0 + 16 + koff);
#pragma unroll
    for (int nt = 0; nt < 4; ++nt) {
      const _Float16* wrow = W + (size_t)(c0 + nt * 16 + ln) * DDIM;
      v16h b = load_frag(wrow + k0 + koff, wrow + k0 + 16 + koff);
      acc[nt] = wmma_f32_f16(a, b, acc[nt]);
    }
  }

#pragma unroll
  for (int nt = 0; nt < 4; ++nt) {
    const int e  = c0 + nt * 16 + ln;
    const float bv = bias[e];
    const int h  = e >> 6;
    const int dk = e & 63;
#pragma unroll
    for (int r = 0; r < 8; ++r) {
      const int t = r0 + r + halfw * 8;         // C-layout row
      float v = acc[nt][r] + bv;
      if (MODE == 3) {
        ((float*)out)[(size_t)t * DDIM + e] = v;
      } else {
        const int b = t >> 11, s = t & (SEQ - 1);
        if (MODE == 0) v *= 0.125f;             // 1/sqrt(DK)
        size_t idx;
        if (MODE == 2) idx = (((size_t)(b * NH + h) * DKD + dk) * SEQ + s);
        else           idx = (((size_t)(b * NH + h) * SEQ + s) * DKD + dk);
        ((_Float16*)out)[idx] = (_Float16)v;
      }
    }
  }
}

// ---------------------------------------------------------------------------
// Flash attention, transposed formulation.
//   S^T = K @ Q^T  : A = K rows (kv x dk), B = Q^T (lane = query column)
//   O^T = V^T @ P^T: A = Vt rows (dk x kv), B = P^T = exp(S^T - m) (in regs!)
// One wave = 16 queries of one (b,h); block = 4 waves; kv tiles of 32.
// C-layout of S^T: lane ln owns query q0+ln; holds kv {0..7}(s0),{16..23}(s1);
// lane ln+16 holds kv {8..15},{24..31}. So softmax stats are per-lane scalars
// plus a single shfl_xor(16), and P^T is exactly the next B-fragment.
// ---------------------------------------------------------------------------
__global__ __launch_bounds__(128) void flash_attn(
    const _Float16* __restrict__ Qh,  // [BH][S][64], pre-scaled by 1/8
    const _Float16* __restrict__ Kh,  // [BH][S][64]
    const _Float16* __restrict__ Vt,  // [BH][64][S]
    _Float16* __restrict__ Oh) {      // [t][1024]
  const int lane  = threadIdx.x & 31;
  const int wave  = threadIdx.x >> 5;
  const int halfw = lane >> 4;
  const int ln    = lane & 15;
  const int koff  = halfw * 8;
  const int bh    = blockIdx.y;                    // 0..BH-1
  const int q0    = blockIdx.x * 64 + wave * 16;
  const size_t base = (size_t)bh * SEQ * DKD;

  // Q^T B-fragments (this lane's query column), loaded once.
  const _Float16* qrow = Qh + base + (size_t)(q0 + ln) * DKD;
  const v16h qb0 = load_frag(qrow + koff,      qrow + 16 + koff);  // dk 0..31
  const v16h qb1 = load_frag(qrow + 32 + koff, qrow + 48 + koff);  // dk 32..63

  float m = -INFINITY;  // running max for query q0+ln (same in both halves)
  float l = 0.0f;       // running denom
  v8f o[4] = {};        // O^T: o[nt] covers dk = nt*16 + r + 8*halfw, col = q

  for (int kv = 0; kv < SEQ; kv += 32) {
    // ---- S^T tile (32 kv x 16 q): two M-tiles, dk split 0..31 / 32..63 ----
    v8f s0 = {}, s1 = {};
    {
      const _Float16* k0r = Kh + base + (size_t)(kv + ln) * DKD;
      v16h ka0 = load_frag(k0r + koff,      k0r + 16 + koff);
      v16h ka1 = load_frag(k0r + 32 + koff, k0r + 48 + koff);
      s0 = wmma_f32_f16(ka0, qb0, s0);
      s0 = wmma_f32_f16(ka1, qb1, s0);
      const _Float16* k1r = Kh + base + (size_t)(kv + 16 + ln) * DKD;
      v16h kb0 = load_frag(k1r + koff,      k1r + 16 + koff);
      v16h kb1 = load_frag(k1r + 32 + koff, k1r + 48 + koff);
      s1 = wmma_f32_f16(kb0, qb0, s1);
      s1 = wmma_f32_f16(kb1, qb1, s1);
      // Prefetch next kv tile's K lines into cache while WMMAs run
      // (speculative; overruns past the tensor end are silently dropped).
      __builtin_prefetch(k0r + 32 * DKD, 0, 3);
      __builtin_prefetch(k1r + 32 * DKD, 0, 3);
    }

    // ---- online softmax: per-lane over 16 elems + one half-swap ----
    float mx = s0[0];
#pragma unroll
    for (int i = 1; i < 8; ++i) mx = fmaxf(mx, s0[i]);
#pragma unroll
    for (int i = 0; i < 8; ++i) mx = fmaxf(mx, s1[i]);
    mx = fmaxf(mx, __shfl_xor(mx, 16, 32));       // combine kv halves
    const float mn    = fmaxf(m, mx);
    const float scale = __expf(m - mn);
    m = mn;

    // P^T = exp(S^T - m): element e of B-fragment is exactly kv
    // (e>=8?16:0)+koff+(e&7), i.e. pb[i]=p(s0[i]), pb[8+i]=p(s1[i]).
    v16h pb;
    float rs = 0.0f;
#pragma unroll
    for (int i = 0; i < 8; ++i) {
      float p0 = __expf(s0[i] - mn);
      float p1 = __expf(s1[i] - mn);
      rs += p0 + p1;
      pb[i]     = (_Float16)p0;
      pb[8 + i] = (_Float16)p1;
    }
    rs += __shfl_xor(rs, 16, 32);                 // combine kv halves
    l = l * scale + rs;

    // ---- rescale running output, then O^T += V^T @ P^T ----
#pragma unroll
    for (int nt = 0; nt < 4; ++nt) {
#pragma unroll
      for (int r = 0; r < 8; ++r) o[nt][r] *= scale;
      const _Float16* vrow = Vt + base + (size_t)(nt * 16 + ln) * SEQ + kv;
      v16h va = load_frag(vrow + koff, vrow + 16 + koff);
      __builtin_prefetch(vrow + 32, 0, 3);        // next kv tile of this dk row
      o[nt] = wmma_f32_f16(va, pb, o[nt]);
    }
  }

  // ---- finalize: divide by l; one b128 store per nt ----
  // C-layout: dk = nt*16 + r + 8*halfw, so the r=0..7 elements are contiguous
  // at column offset h*64 + nt*16 + 8*halfw (= +koff).
  const int b = bh >> 4, h = bh & 15;
  const int t = b * SEQ + q0 + ln;
  const float inv_l = 1.0f / l;
  _Float16* orow = Oh + (size_t)t * DDIM + h * 64 + koff;
#pragma unroll
  for (int nt = 0; nt < 4; ++nt) {
    v8h vout;
#pragma unroll
    for (int r = 0; r < 8; ++r) vout[r] = (_Float16)(o[nt][r] * inv_l);
    *(v8h*)(orow + nt * 16) = vout;
  }
}

// ---------------------------------------------------------------------------
// Host-side orchestration. Workspace layout (48 MB):
//   [0,8M)    x_h   | [8M,10M) wq_h | [10M,12M) wk_h | [12M,14M) wv_h
//   [14M,16M) wo_h  | [16M,24M) Q_h [B,H,S,DK] | [24M,32M) K_h [B,H,S,DK]
//   [32M,40M) Vt_h [B,H,DK,S] | [40M,48M) attn_h [t][1024]
// ---------------------------------------------------------------------------
extern "C" void kernel_launch(void* const* d_in, const int* in_sizes, int n_in,
                              void* d_out, int out_size, void* d_ws, size_t ws_size,
                              hipStream_t stream) {
  const float* x  = (const float*)d_in[0];
  const float* wq = (const float*)d_in[1];
  const float* bq = (const float*)d_in[2];
  const float* wk = (const float*)d_in[3];
  const float* bk = (const float*)d_in[4];
  const float* wv = (const float*)d_in[5];
  const float* bv = (const float*)d_in[6];
  const float* wo = (const float*)d_in[7];
  const float* bo = (const float*)d_in[8];

  char* ws = (char*)d_ws;
  const size_t MB = 1024 * 1024;
  _Float16* x_h    = (_Float16*)(ws + 0 * MB);
  _Float16* wq_h   = (_Float16*)(ws + 8 * MB);
  _Float16* wk_h   = (_Float16*)(ws + 10 * MB);
  _Float16* wv_h   = (_Float16*)(ws + 12 * MB);
  _Float16* wo_h   = (_Float16*)(ws + 14 * MB);
  _Float16* Q_h    = (_Float16*)(ws + 16 * MB);
  _Float16* K_h    = (_Float16*)(ws + 24 * MB);
  _Float16* Vt_h   = (_Float16*)(ws + 32 * MB);
  _Float16* attn_h = (_Float16*)(ws + 40 * MB);

  // 1) fp32 -> f16 conversions
  const int nX4 = (MTOT * DDIM) / 4;      // 1M
  const int nW4 = (DDIM * DDIM) / 4;      // 256K
  cvt_f32_f16<<<2048, 256, 0, stream>>>(x,  x_h,  nX4);
  cvt_f32_f16<<<1024, 256, 0, stream>>>(wq, wq_h, nW4);
  cvt_f32_f16<<<1024, 256, 0, stream>>>(wk, wk_h, nW4);
  cvt_f32_f16<<<1024, 256, 0, stream>>>(wv, wv_h, nW4);
  cvt_f32_f16<<<1024, 256, 0, stream>>>(wo, wo_h, nW4);

  // 2) QKV projections (WMMA GEMM, 64x64 tile per block of 4 waves)
  dim3 ggrid(MTOT / 64, DDIM / 64);       // (64, 16)
  dim3 gblk(128);
  gemm_wmma<0><<<ggrid, gblk, 0, stream>>>(x_h, wq_h, bq, (void*)Q_h);
  gemm_wmma<1><<<ggrid, gblk, 0, stream>>>(x_h, wk_h, bk, (void*)K_h);
  gemm_wmma<2><<<ggrid, gblk, 0, stream>>>(x_h, wv_h, bv, (void*)Vt_h);

  // 3) flash attention: grid = (S/64 q-tiles, B*H), 4 waves/block
  dim3 agrid(SEQ / 64, NB * NH);          // (32, 32)
  flash_attn<<<agrid, gblk, 0, stream>>>(Q_h, K_h, Vt_h, attn_h);

  // 4) output projection -> fp32 d_out
  gemm_wmma<3><<<ggrid, gblk, 0, stream>>>(attn_h, wo_h, bo, d_out);
}
